// MambaBlock_66546223284397
// MI455X (gfx1250) — matrix operand order repaired and checked by reading, not compiled, not run on hardware
//
#include <hip/hip_runtime.h>

// ---------------- problem constants ----------------
#define D_MODEL 1024
#define D_STATE 16
#define D_CONV  4
#define D_INNER 2048            // EXPAND * D_MODEL
#define BATCH   2
#define SEQ     1024
#define NTOK    (BATCH * SEQ)   // 2048 tokens

// ---------------- vector / WMMA types ----------------
typedef __attribute__((ext_vector_type(16))) __bf16        v16bf;
typedef __attribute__((ext_vector_type(8)))  float         v8f;
typedef __attribute__((ext_vector_type(4)))  unsigned int  uint4v;
typedef __attribute__((ext_vector_type(2)))  unsigned int  uint2v;
typedef __attribute__((ext_vector_type(4)))  unsigned int  u32x4;
typedef __attribute__((ext_vector_type(8)))  unsigned int  u32x8;

union Frag32 {            // 32 bytes = one bf16 WMMA A/B fragment per lane
    uint4v q[2];
    v16bf  v;
};

__device__ __forceinline__ unsigned short f2bf(float f) {
    // round-to-nearest-even f32 -> bf16
    unsigned int u = __float_as_uint(f);
    u += 0x7FFFu + ((u >> 16) & 1u);
    return (unsigned short)(u >> 16);
}

__device__ __forceinline__ unsigned int pack_bf2(float lo, float hi) {
    return (unsigned int)f2bf(lo) | ((unsigned int)f2bf(hi) << 16);
}

__device__ __forceinline__ float silu(float x) {
    return x / (1.0f + __expf(-x));
}

// =====================================================================
// f32 -> bf16 elementwise (8 elements / thread, b128 in, b128 out)
// =====================================================================
__global__ __launch_bounds__(256)
void f32_to_bf16_kernel(const float* __restrict__ src,
                        unsigned short* __restrict__ dst, int n8)
{
    int i = blockIdx.x * blockDim.x + threadIdx.x;
    if (i >= n8) return;
    float4 a = ((const float4*)src)[2 * i];
    float4 b = ((const float4*)src)[2 * i + 1];
    uint4v o = { pack_bf2(a.x, a.y), pack_bf2(a.z, a.w),
                 pack_bf2(b.x, b.y), pack_bf2(b.z, b.w) };
    ((uint4v*)dst)[i] = o;
}

// =====================================================================
// f32 [K][N] -> bf16 [N][K] transpose-convert, 32x32 tiles through LDS.
// =====================================================================
#define TT 32
__global__ __launch_bounds__(256)
void transpose_bf16_kernel(const float* __restrict__ src,
                           unsigned short* __restrict__ dst,
                           int K, int N)
{
    __shared__ __align__(16) unsigned short tile[TT][TT + 4]; // 72B rows
    const int tid = threadIdx.x;
    const int k0 = blockIdx.y * TT;
    const int n0 = blockIdx.x * TT;
    const int r  = tid >> 3;          // 0..31
    const int c4 = (tid & 7) * 4;     // 0..28 step 4

    float4 v = *(const float4*)&src[(size_t)(k0 + r) * N + n0 + c4];
    tile[c4 + 0][r] = f2bf(v.x);
    tile[c4 + 1][r] = f2bf(v.y);
    tile[c4 + 2][r] = f2bf(v.z);
    tile[c4 + 3][r] = f2bf(v.w);
    __syncthreads();

    uint2v o = *(const uint2v*)&tile[r][c4];
    *(uint2v*)&dst[(size_t)(n0 + r) * K + k0 + c4] = o;
}

// =====================================================================
// Tensor Data Mover: issue a 2-D bf16 tile load (tile_dim0=32 elems,
// tile_dim1=128 rows) into LDS at lds_off, with LDS padding of 4 DWORDs
// every 16 DWORDs (row stride 80B = BK + PADK bf16 elements).
// D# built per CDNA5 ISA 8.3/8.4; groups 2/3 omitted (2-D tensor).
// =====================================================================
#define BM 128
#define BN 128
#define BK 32
#define PADK 8            // bf16 elements of LDS row padding (16B)
#define LROW (BK + PADK)  // 40 bf16 elements = 80B LDS row stride
#define TILE_BYTES (BM * LROW * 2)            // 10240 B per operand tile
#define BUF_BYTES  (2 * TILE_BYTES)           // lA + lB per buffer

__device__ __forceinline__ void tdm_load_tile_2d(unsigned lds_off,
                                                 const unsigned short* gsrc,
                                                 unsigned tensor_d0,
                                                 unsigned tensor_d1,
                                                 unsigned stride_d0)
{
    unsigned long long ga = (unsigned long long)gsrc;
    u32x4 g0;
    g0[0] = 1u;                                          // count=1, load, user
    g0[1] = lds_off;                                     // lds_addr (bytes)
    g0[2] = (unsigned)(ga & 0xFFFFFFFFu);                // global_addr[31:0]
    g0[3] = (unsigned)((ga >> 32) & 0x01FFFFFFu)         // global_addr[56:32]
          | (2u << 30);                                  // type=2 (image)
    u32x8 g1;
    g1[0] = (1u << 16)                                   // data_size=1 (2B)
          | (1u << 20)                                   // pad_enable
          | (3u << 22)                                   // pad_interval: 16 DW
          | (3u << 25);                                  // pad_amount: 4 DW
    g1[1] = (tensor_d0 & 0xFFFFu) << 16;                 // tensor_dim0[15:0]
    g1[2] = (tensor_d0 >> 16) | ((tensor_d1 & 0xFFFFu) << 16);
    g1[3] = (tensor_d1 >> 16) | ((unsigned)BK << 16);    // tile_dim0=32
    g1[4] = (unsigned)BM;                                // tile_dim1=128
    g1[5] = stride_d0;                                   // dim0_stride[31:0]
    g1[6] = 0;                                           // stride hi, dim1_stride lo
    g1[7] = 0;
    asm volatile("tensor_load_to_lds %0, %1"
                 :: "s"(g0), "s"(g1)
                 : "memory");
}

// =====================================================================
// Tiled GEMM with bias:  C[M,N] = A[M,K] @ Bt[N,K]^T + bias[N]
// A, Bt bf16.  Block tile 128x128, K-step 32, 256 threads = 8 wave32
// waves; wave (wmr 0..1, wnc 0..3) owns 64x32 = 4x2 accumulators ->
// 8 v_wmma_f32_16x16x32_bf16 per wave per K-step.  K-panels staged by
// the Tensor Data Mover, double-buffered, synced with s_wait_tensorcnt.
// =====================================================================
__global__ __launch_bounds__(256)
void gemm_bf16_wmma_tdm(const unsigned short* __restrict__ A,   // [M][K]
                        const unsigned short* __restrict__ Bt,  // [N][K]
                        const float* __restrict__ bias,
                        float* __restrict__ C,
                        int M, int N, int K)
{
    extern __shared__ unsigned short smem[];   // 2 buffers x (lA | lB)

    const int tid  = threadIdx.x;
    const int lane = tid & 31;
    const int wave = tid >> 5;
    const int wmr  = wave & 1;        // 0..1 -> 64-row half
    const int wnc  = wave >> 1;       // 0..3 -> 32-col strip
    const int hl   = lane & 15;
    const int hi   = lane >> 4;

    const int bm = blockIdx.y * BM;
    const int bn = blockIdx.x * BN;

    v8f acc[4][2];
#pragma unroll
    for (int mt = 0; mt < 4; ++mt)
#pragma unroll
        for (int nt = 0; nt < 2; ++nt)
            acc[mt][nt] = 0.0f;

    const unsigned short* Abase  = A  + (size_t)bm * K;
    const unsigned short* Btbase = Bt + (size_t)bn * K;
    const int nk = K / BK;

    // prologue: stage panel 0 into buffer 0
    if (wave == 0) {
        tdm_load_tile_2d(0,          Abase,  K, M, K);
        tdm_load_tile_2d(TILE_BYTES, Btbase, K, N, K);
    }

    for (int i = 0; i < nk; ++i) {
        const int cur = i & 1;
        if (wave == 0) {
            if (i + 1 < nk) {
                const int ko2  = (i + 1) * BK;
                const unsigned nb = (unsigned)(cur ^ 1) * BUF_BYTES;
                tdm_load_tile_2d(nb,              Abase  + ko2, K, M, K);
                tdm_load_tile_2d(nb + TILE_BYTES, Btbase + ko2, K, N, K);
                __builtin_amdgcn_s_wait_tensorcnt(2);  // panel i complete
            } else {
                __builtin_amdgcn_s_wait_tensorcnt(0);
            }
        }
        __syncthreads();   // panel i visible to all waves

        const unsigned short* lA = smem + (size_t)cur * (BUF_BYTES / 2);
        const unsigned short* lB = lA + TILE_BYTES / 2;

        // A fragments: row = wmr*64 + mt*16 + hl; K chunks {hi*8, 16+hi*8}
        Frag32 af[4];
#pragma unroll
        for (int mt = 0; mt < 4; ++mt) {
            const int row = wmr * 64 + mt * 16 + hl;
            af[mt].q[0] = *(const uint4v*)&lA[row * LROW + hi * 8];
            af[mt].q[1] = *(const uint4v*)&lA[row * LROW + 16 + hi * 8];
        }
        // B fragments: col = wnc*32 + nt*16 + hl; K chunks {hi*8, 16+hi*8}
        Frag32 bf[2];
#pragma unroll
        for (int nt = 0; nt < 2; ++nt) {
            const int col = wnc * 32 + nt * 16 + hl;
            bf[nt].q[0] = *(const uint4v*)&lB[col * LROW + hi * 8];
            bf[nt].q[1] = *(const uint4v*)&lB[col * LROW + 16 + hi * 8];
        }
#pragma unroll
        for (int mt = 0; mt < 4; ++mt)
#pragma unroll
            for (int nt = 0; nt < 2; ++nt)
                acc[mt][nt] = __builtin_amdgcn_wmma_f32_16x16x32_bf16(
                    false, af[mt].v, false, bf[nt].v, (short)0,
                    acc[mt][nt], false, false);
        __syncthreads();   // buffer consumed before TDM overwrites it
    }

    // epilogue: D layout -> lane holds (M = hi*8 + r, N = hl)
#pragma unroll
    for (int nt = 0; nt < 2; ++nt) {
        const int col = bn + wnc * 32 + nt * 16 + hl;
        const float bv = bias[col];
#pragma unroll
        for (int mt = 0; mt < 4; ++mt) {
            const int rowb = bm + wmr * 64 + mt * 16 + hi * 8;
#pragma unroll
            for (int r = 0; r < 8; ++r)
                C[(size_t)(rowb + r) * N + col] = acc[mt][nt][r] + bv;
        }
    }
}

// =====================================================================
// Depthwise causal conv (width 4) + bias + SiLU.
// =====================================================================
__global__ __launch_bounds__(256)
void conv_silu_kernel(const float* __restrict__ xr,
                      const float* __restrict__ conv_w,
                      const float* __restrict__ conv_b,
                      float* __restrict__ xc)
{
    int idx = blockIdx.x * blockDim.x + threadIdx.x;
    if (idx >= NTOK * D_INNER) return;
    int d  = idx & (D_INNER - 1);
    int tl = idx >> 11;                 // token-linear index (b*SEQ + t)
    int t  = tl & (SEQ - 1);

    float acc = conv_b[d];
#pragma unroll
    for (int k = 0; k < D_CONV; ++k) {
        int tt = t - (D_CONV - 1) + k;
        if (tt >= 0)
            acc += xr[(size_t)(tl - (D_CONV - 1) + k) * (2 * D_INNER) + d] *
                   conv_w[k * D_INNER + d];
    }
    xc[(size_t)tl * D_INNER + d] = silu(acc);
}

// =====================================================================
// x_proj: proj[NTOK, 48] = xc @ x_proj_w + b   (too skinny for WMMA)
// =====================================================================
__global__ __launch_bounds__(256)
void xproj_kernel(const float* __restrict__ xc,
                  const float* __restrict__ w,
                  const float* __restrict__ b,
                  float* __restrict__ proj)
{
    int idx = blockIdx.x * blockDim.x + threadIdx.x;
    if (idx >= NTOK * 3 * D_STATE) return;
    int j  = idx % (3 * D_STATE);
    int tl = idx / (3 * D_STATE);

    const float* xrow = xc + (size_t)tl * D_INNER;
    float acc = b[j];
    for (int k = 0; k < D_INNER; ++k)
        acc += xrow[k] * w[(size_t)k * (3 * D_STATE) + j];
    proj[(size_t)tl * (3 * D_STATE) + j] = acc;
}

// =====================================================================
// Fused dt_proj -> softplus -> selective scan -> D-skip -> SiLU gate.
// One (batch, channel) per lane; h[16] in VGPRs; emits gated as bf16.
// =====================================================================
__global__ __launch_bounds__(256)
void scan_kernel(const float* __restrict__ xc,
                 const float* __restrict__ xr,      // res = xr[:, D_INNER:]
                 const float* __restrict__ proj,    // [NTOK, 48] = d_raw|B|C
                 const float* __restrict__ dt_w,    // [D_STATE, D_INNER]
                 const float* __restrict__ dt_b,    // [D_INNER]
                 const float* __restrict__ A_log,   // [D_INNER, D_STATE]
                 const float* __restrict__ D_skip,  // [D_INNER]
                 unsigned short* __restrict__ gated_bf)
{
    int ch = blockIdx.x * blockDim.x + threadIdx.x;
    if (ch >= BATCH * D_INNER) return;
    int b = ch >> 11;               // / D_INNER
    int d = ch & (D_INNER - 1);

    float Acoef[D_STATE], dtw[D_STATE], h[D_STATE];
#pragma unroll
    for (int n = 0; n < D_STATE; ++n) {
        Acoef[n] = -__expf(A_log[(size_t)d * D_STATE + n]);
        dtw[n]   = dt_w[(size_t)n * D_INNER + d];
        h[n]     = 0.0f;
    }
    const float dtb = dt_b[d];
    const float Dd  = D_skip[d];

    for (int t = 0; t < SEQ; ++t) {
        int tl = b * SEQ + t;
        const float* pr = proj + (size_t)tl * (3 * D_STATE);

        float dacc = dtb;
#pragma unroll
        for (int n = 0; n < D_STATE; ++n) dacc += pr[n] * dtw[n];
        float delta = (dacc > 20.0f) ? dacc : __logf(1.0f + __expf(dacc));

        float x_t = xc[(size_t)tl * D_INNER + d];
        float dx  = delta * x_t;

        float y = 0.0f;
#pragma unroll
        for (int n = 0; n < D_STATE; ++n) {
            float dA = __expf(delta * Acoef[n]);
            h[n] = dA * h[n] + dx * pr[D_STATE + n];        // B
            y   += h[n] * pr[2 * D_STATE + n];              // C
        }
        float yv = y + x_t * Dd;
        float r  = xr[(size_t)tl * (2 * D_INNER) + D_INNER + d];
        gated_bf[(size_t)tl * D_INNER + d] = f2bf(yv * silu(r));
    }
}

// =====================================================================
// Launch
// =====================================================================
extern "C" void kernel_launch(void* const* d_in, const int* in_sizes, int n_in,
                              void* d_out, int out_size, void* d_ws, size_t ws_size,
                              hipStream_t stream) {
    (void)in_sizes; (void)n_in; (void)out_size; (void)ws_size;

    const float* x         = (const float*)d_in[0];
    const float* in_proj_w = (const float*)d_in[1];
    const float* in_proj_b = (const float*)d_in[2];
    const float* conv_w    = (const float*)d_in[3];
    const float* conv_b    = (const float*)d_in[4];
    const float* x_proj_w  = (const float*)d_in[5];
    const float* x_proj_b  = (const float*)d_in[6];
    const float* dt_proj_w = (const float*)d_in[7];
    const float* dt_proj_b = (const float*)d_in[8];
    const float* A_log     = (const float*)d_in[9];
    const float* D_skip    = (const float*)d_in[10];
    const float* out_proj_w= (const float*)d_in[11];
    const float* out_proj_b= (const float*)d_in[12];
    float*       out       = (float*)d_out;

    // ---- workspace layout ----
    float* ws   = (float*)d_ws;
    float* xr   = ws;                                   // NTOK*4096 f32
    float* xc   = xr + (size_t)NTOK * 2 * D_INNER;      // NTOK*2048 f32
    float* proj = xc + (size_t)NTOK * D_INNER;          // NTOK*48   f32
    unsigned short* xb    = (unsigned short*)(proj + (size_t)NTOK * 3 * D_STATE);
    unsigned short* wint  = xb    + (size_t)NTOK * D_MODEL;          // 4096x1024 bf16 (W_in^T)
    unsigned short* woutt = wint  + (size_t)(2 * D_INNER) * D_MODEL; // 1024x2048 bf16 (W_out^T)
    unsigned short* gbf   = woutt + (size_t)D_MODEL * D_INNER;       // NTOK*2048 bf16

    // 0) bf16 conversions / weight transposes
    f32_to_bf16_kernel<<<(NTOK * D_MODEL / 8 + 255) / 256, 256, 0, stream>>>(
        x, xb, NTOK * D_MODEL / 8);
    transpose_bf16_kernel<<<dim3((2 * D_INNER) / TT, D_MODEL / TT), 256, 0, stream>>>(
        in_proj_w, wint, D_MODEL, 2 * D_INNER);
    transpose_bf16_kernel<<<dim3(D_MODEL / TT, D_INNER / TT), 256, 0, stream>>>(
        out_proj_w, woutt, D_INNER, D_MODEL);

    // 1) in_proj: xr = x @ W_in + b_in   (M=2048, N=4096, K=1024)
    gemm_bf16_wmma_tdm<<<dim3((2 * D_INNER) / BN, NTOK / BM), 256,
                         2 * BUF_BYTES, stream>>>(
        xb, wint, in_proj_b, xr, NTOK, 2 * D_INNER, D_MODEL);

    // 2) depthwise conv + SiLU
    conv_silu_kernel<<<(NTOK * D_INNER) / 256, 256, 0, stream>>>(
        xr, conv_w, conv_b, xc);

    // 3) x_proj
    xproj_kernel<<<(NTOK * 3 * D_STATE + 255) / 256, 256, 0, stream>>>(
        xc, x_proj_w, x_proj_b, proj);

    // 4) fused dt_proj + softplus + selective scan + gate  -> bf16
    scan_kernel<<<(BATCH * D_INNER) / 256, 256, 0, stream>>>(
        xc, xr, proj, dt_proj_w, dt_proj_b, A_log, D_skip, gbf);

    // 5) out_proj: out = gated @ W_out + b_out   (M=2048, N=1024, K=2048)
    gemm_bf16_wmma_tdm<<<dim3(D_MODEL / BN, NTOK / BM), 256,
                         2 * BUF_BYTES, stream>>>(
        gbf, woutt, out_proj_b, out, NTOK, D_MODEL, D_INNER);
}
